// GraphEncoder_45140106281146
// MI455X (gfx1250) — matrix operand order repaired and checked by reading, not compiled, not run on hardware
//
#include <hip/hip_runtime.h>
#include <hip/hip_bf16.h>
#include <math.h>

typedef __bf16 bf16;
typedef __attribute__((ext_vector_type(8)))  bf16  v8bf;
typedef __attribute__((ext_vector_type(16))) bf16  v16bf;
typedef __attribute__((ext_vector_type(8)))  float v8f;
typedef __attribute__((ext_vector_type(4)))  int   v4i;

#define HID   128
#define INCH  64

// ---------------------------------------------------------------- utilities
__global__ void k_fill_f32(float* __restrict__ p, long n, float v) {
  long i = (long)blockIdx.x * blockDim.x + threadIdx.x;
  long st = (long)gridDim.x * blockDim.x;
  for (; i < n; i += st) p[i] = v;
}

__global__ void k_deg_accum(float* __restrict__ deg, const int* __restrict__ dst,
                            const float* __restrict__ w, int E) {
  int i = blockIdx.x * blockDim.x + threadIdx.x;
  int st = gridDim.x * blockDim.x;
  for (; i < E; i += st) atomicAdd(&deg[dst[i]], w[i]);
}

__global__ void k_rsqrt_inplace(float* __restrict__ d, int N) {
  int i = blockIdx.x * blockDim.x + threadIdx.x;
  int st = gridDim.x * blockDim.x;
  for (; i < N; i += st) {
    float v = d[i];
    d[i] = (v > 0.0f) ? rsqrtf(v) : 0.0f;
  }
}

// norm[e] = dinv[src]*w[e]*dinv[dst] ; precomputed once, reused by 3 layers
__global__ void k_enorm(const int* __restrict__ ei, const float* __restrict__ w,
                        const float* __restrict__ dinv, float* __restrict__ enorm, int E) {
  int i = blockIdx.x * blockDim.x + threadIdx.x;
  int st = gridDim.x * blockDim.x;
  for (; i < E; i += st) {
    int s = ei[i];
    int d = ei[E + i];
    enorm[i] = dinv[s] * w[i] * dinv[d];
  }
}

__global__ void k_f2bf(const float* __restrict__ in, bf16* __restrict__ out, long n) {
  long i = (long)blockIdx.x * blockDim.x + threadIdx.x;
  long st = (long)gridDim.x * blockDim.x;
  for (; i < n; i += st) out[i] = (bf16)in[i];
}

// W [Cin][128] fp32  ->  Wt [128][Cin] bf16 (row n holds column n of W)
__global__ void k_wtrans(const float* __restrict__ W, bf16* __restrict__ Wt, int Cin) {
  int i = blockIdx.x * blockDim.x + threadIdx.x;
  int total = Cin * HID;
  if (i >= total) return;
  int n = i / Cin;
  int k = i - n * Cin;
  Wt[i] = (bf16)W[(long)k * HID + n];
}

// ---------------------------------------------------------------- WMMA GEMM
// C[N,128] = A[N,Cin](bf16) x W[Cin,128], W given transposed as Wt[128][Cin] bf16.
// 256 threads = 8 waves; each wave computes a 16-row strip across all 8 col tiles.
// Wt staged in LDS once per block (async-to-LDS when available). Fragment layouts
// used by this code (contiguous K order, not the ISA table order):
//   A 16x32 bf16 : lanes 0-15 -> m=lane,K[0..15]  ; lanes 16-31 -> m=lane-16,K[16..31]
//   B 32x16 bf16 : lanes 0-15 -> n=lane,K[0..15]  ; lanes 16-31 -> n=lane-16,K[16..31]
//   C 16x16 f32  : VGPR r, lanes 0-15 -> M=r ; lanes 16-31 -> M=8+r
__global__ void __launch_bounds__(256)
k_gemm_wmma(const bf16* __restrict__ A, const bf16* __restrict__ Wt,
            float* __restrict__ C, int numRowTiles, int Cin) {
  extern __shared__ char smem[];
  bf16* lds = (bf16*)smem;

  // stage Wt (128*Cin bf16) into LDS, 16B chunks
  {
    int nchunks = (HID * Cin * 2) >> 4;
    const uint4* src = (const uint4*)Wt;
    uint4* dst = (uint4*)smem;
#if __has_builtin(__builtin_amdgcn_global_load_async_to_lds_b128)
    for (int i = threadIdx.x; i < nchunks; i += blockDim.x) {
      __builtin_amdgcn_global_load_async_to_lds_b128(
          (__attribute__((address_space(1))) v4i*)const_cast<uint4*>(src + i),
          (__attribute__((address_space(3))) v4i*)(dst + i), 0, 0);
    }
#if __has_builtin(__builtin_amdgcn_s_wait_asynccnt)
    __builtin_amdgcn_s_wait_asynccnt(0);
#else
    asm volatile("s_wait_asynccnt 0x0" ::: "memory");
#endif
#else
    for (int i = threadIdx.x; i < nchunks; i += blockDim.x) dst[i] = src[i];
#endif
  }
  __syncthreads();

  int wave = threadIdx.x >> 5;
  int lane = threadIdx.x & 31;
  int rt   = blockIdx.x * (blockDim.x >> 5) + wave;
  if (rt >= numRowTiles) return;

  int m    = lane & 15;
  int half = lane >> 4;
  int ksteps = Cin >> 5;

  v8f acc[8];
#pragma unroll
  for (int nt = 0; nt < 8; ++nt)
#pragma unroll
    for (int r = 0; r < 8; ++r) acc[nt][r] = 0.0f;

  const bf16* arow = A + (size_t)(rt * 16 + m) * Cin;

  for (int ks = 0; ks < ksteps; ++ks) {
    // A fragment: two 16B chunks per lane (contiguous K-half)
    v8bf alo = *(const v8bf*)(arow + ks * 32 + half * 16);
    v8bf ahi = *(const v8bf*)(arow + ks * 32 + half * 16 + 8);
    v16bf a = __builtin_shufflevector(alo, ahi, 0, 1, 2, 3, 4, 5, 6, 7,
                                      8, 9, 10, 11, 12, 13, 14, 15);
    // batch-load all 8 B fragments (one wait), then 8 back-to-back WMMAs
    v16bf b[8];
#pragma unroll
    for (int nt = 0; nt < 8; ++nt)
      b[nt] = *(const v16bf*)(lds + (size_t)(nt * 16 + m) * Cin + ks * 32 + half * 16);
#pragma unroll
    for (int nt = 0; nt < 8; ++nt)
      acc[nt] = __builtin_amdgcn_wmma_f32_16x16x32_bf16(
          false, a, false, b[nt], (short)0, acc[nt], false, false);
  }

  float* crow = C + (size_t)(rt * 16) * HID;
#pragma unroll
  for (int nt = 0; nt < 8; ++nt) {
    int col = nt * 16 + m;
#pragma unroll
    for (int r = 0; r < 8; ++r)
      crow[(size_t)(r + 8 * half) * HID + col] = acc[nt][r];
  }
}

// ---------------------------------------------------------------- scatter-add
// one wave per edge; lane owns 4 channels (float4 gather, 4 f32 atomics).
// Next iteration's gather row is prefetched (global_prefetch_b8) to hide the
// dependent index->gather latency behind the current edge's atomics.
__global__ void __launch_bounds__(256)
k_scatter(const float* __restrict__ hw, const int* __restrict__ ei,
          const float* __restrict__ enorm, float* __restrict__ agg, int E) {
  int gwave = (int)((blockIdx.x * blockDim.x + threadIdx.x) >> 5);
  int nwave = (int)((gridDim.x * blockDim.x) >> 5);
  int lane  = threadIdx.x & 31;
  for (int e = gwave; e < E; e += nwave) {
    int s = ei[e];
    int d = ei[E + e];
    float nm = enorm[e];
    int enext = e + nwave;
    if (enext < E) {
      int snext = ei[enext];
      __builtin_prefetch(hw + (size_t)snext * HID + lane * 4, 0, 3);
    }
    float4 v = ((const float4*)(hw + (size_t)s * HID))[lane];
    float* q = agg + (size_t)d * HID + lane * 4;
    atomicAdd(q + 0, v.x * nm);
    atomicAdd(q + 1, v.y * nm);
    atomicAdd(q + 2, v.z * nm);
    atomicAdd(q + 3, v.w * nm);
  }
}

// bias + self-loop term + SiLU, re-encode bf16 for next GEMM
__global__ void k_act(const float* __restrict__ agg, const float* __restrict__ hw,
                      const float* __restrict__ dinv, const float* __restrict__ bias,
                      bf16* __restrict__ actbf, long total) {
  long i = (long)blockIdx.x * blockDim.x + threadIdx.x;
  long st = (long)gridDim.x * blockDim.x;
  for (; i < total; i += st) {
    int  c    = (int)(i & (HID - 1));
    long node = i >> 7;
    float di  = dinv[node];
    float val = agg[i] + bias[c] + di * di * hw[i];
    float s   = val * (1.0f / (1.0f + expf(-val)));
    actbf[i]  = (bf16)s;
  }
}

// global mean pool: per-block partial sums then one atomic per channel
__global__ void k_reduce(const bf16* __restrict__ actbf, float* __restrict__ out,
                         int N, float invN) {
  __shared__ float sm[256];
  int t = threadIdx.x;
  int c = t & (HID - 1);
  int sub = t >> 7;  // 0 or 1
  float sum = 0.0f;
  for (int i = blockIdx.x * 2 + sub; i < N; i += gridDim.x * 2)
    sum += (float)actbf[(size_t)i * HID + c];
  sm[t] = sum;
  __syncthreads();
  if (t < HID) atomicAdd(&out[c], (sm[t] + sm[t + HID]) * invN);
}

// ---------------------------------------------------------------- launcher
extern "C" void kernel_launch(void* const* d_in, const int* in_sizes, int n_in,
                              void* d_out, int out_size, void* d_ws, size_t ws_size,
                              hipStream_t stream) {
  const float* x  = (const float*)d_in[0];
  const int*   ei = (const int*)d_in[1];
  const float* ew = (const float*)d_in[2];
  const float* W1 = (const float*)d_in[3];
  const float* b1 = (const float*)d_in[4];
  const float* W2 = (const float*)d_in[5];
  const float* b2 = (const float*)d_in[6];
  const float* W3 = (const float*)d_in[7];
  const float* b3 = (const float*)d_in[8];
  float* out = (float*)d_out;

  const int N = in_sizes[0] / INCH;
  const int E = in_sizes[1] / 2;

  char* ws = (char*)d_ws;
  size_t off = 0;
  auto alloc = [&](size_t bytes) -> void* {
    void* p = ws + off;
    off = (off + bytes + 255) & ~(size_t)255;
    return p;
  };
  float* dinv  = (float*)alloc((size_t)N * 4);         // degree, then rsqrt in place
  float* enorm = (float*)alloc((size_t)E * 4);
  bf16*  actbf = (bf16*) alloc((size_t)N * HID * 2);   // GEMM input (layer1 uses [N,64])
  float* hw    = (float*)alloc((size_t)N * HID * 4);   // GEMM output (pre-aggregation)
  float* agg   = (float*)alloc((size_t)N * HID * 4);   // scatter destination
  bf16*  W1t   = (bf16*) alloc((size_t)HID * INCH * 2);
  bf16*  W2t   = (bf16*) alloc((size_t)HID * HID * 2);
  bf16*  W3t   = (bf16*) alloc((size_t)HID * HID * 2);
  (void)ws_size; (void)n_in; (void)out_size;

  const int TB = 256;

  // GCN normalization (self-loop weight 1 folded into init value)
  k_fill_f32<<<1024, TB, 0, stream>>>(dinv, N, 1.0f);
  k_deg_accum<<<2048, TB, 0, stream>>>(dinv, ei + E, ew, E);
  k_rsqrt_inplace<<<1024, TB, 0, stream>>>(dinv, N);
  k_enorm<<<2048, TB, 0, stream>>>(ei, ew, dinv, enorm, E);

  // weights -> transposed bf16
  k_wtrans<<<(HID * INCH + TB - 1) / TB, TB, 0, stream>>>(W1, W1t, INCH);
  k_wtrans<<<(HID * HID + TB - 1) / TB, TB, 0, stream>>>(W2, W2t, HID);
  k_wtrans<<<(HID * HID + TB - 1) / TB, TB, 0, stream>>>(W3, W3t, HID);

  // x -> bf16
  k_f2bf<<<2048, TB, 0, stream>>>(x, actbf, (long)N * INCH);

  const int numRT    = N / 16;            // 100000 / 16 = 6250, exact
  const int rtPerBlk = TB / 32;           // 8 waves per block
  const int gblocks  = (numRT + rtPerBlk - 1) / rtPerBlk;

  const float* biases[3] = {b1, b2, b3};
  const bf16*  wts[3]    = {W1t, W2t, W3t};
  const int    cins[3]   = {INCH, HID, HID};

  for (int L = 0; L < 3; ++L) {
    size_t ldsBytes = (size_t)HID * cins[L] * 2;
    k_gemm_wmma<<<gblocks, TB, ldsBytes, stream>>>(actbf, wts[L], hw, numRT, cins[L]);
    k_fill_f32<<<4096, TB, 0, stream>>>(agg, (long)N * HID, 0.0f);
    k_scatter<<<8192, TB, 0, stream>>>(hw, ei, enorm, agg, E);
    k_act<<<8192, TB, 0, stream>>>(agg, hw, dinv, biases[L], actbf, (long)N * HID);
  }

  k_fill_f32<<<1, 128, 0, stream>>>(out, HID, 0.0f);
  k_reduce<<<256, 256, 0, stream>>>(actbf, out, N, 1.0f / (float)N);
}
